// CausalMultiHeadSelfAttention_14267881357937
// MI455X (gfx1250) — matrix operand order repaired and checked by reading, not compiled, not run on hardware
//
#include <hip/hip_runtime.h>
#include <hip/hip_bf16.h>
#include <cmath>

#define D_MODEL   1024
#define NUM_HEADS 16
#define HEAD_DIM  64
#define SEQ       2048
#define BATCH     4
#define T_TOTAL   (BATCH * SEQ)

typedef __attribute__((ext_vector_type(16))) _Float16 v16h;
typedef __attribute__((ext_vector_type(8)))  _Float16 v8h;
typedef __attribute__((ext_vector_type(8)))  float    v8f;
typedef __attribute__((ext_vector_type(4)))  int      v4i;
typedef __attribute__((address_space(1)))    v4i      v4i_g;  // global
typedef __attribute__((address_space(3)))    v4i      v4i_l;  // LDS

#if __has_builtin(__builtin_amdgcn_global_load_async_to_lds_b128)
#define HAS_ASYNC_LDS 1
#else
#define HAS_ASYNC_LDS 0
#endif

// Stage 16 bytes global -> LDS. Async (ASYNCcnt-tracked) when available,
// otherwise plain load + ds_store (made visible by the workgroup barrier).
__device__ __forceinline__ void stage16B(const _Float16* gp, _Float16* lp) {
#if HAS_ASYNC_LDS
  __builtin_amdgcn_global_load_async_to_lds_b128((v4i_g*)gp, (v4i_l*)lp, 0, 0);
#else
  *(v8h*)lp = *(const v8h*)gp;
#endif
}

__device__ __forceinline__ void wait_async_le1() {
#if HAS_ASYNC_LDS
#if __has_builtin(__builtin_amdgcn_s_wait_asynccnt)
  __builtin_amdgcn_s_wait_asynccnt(1);
#else
  asm volatile("s_wait_asynccnt 1" ::: "memory");
#endif
#endif
}
__device__ __forceinline__ void wait_async_le0() {
#if HAS_ASYNC_LDS
#if __has_builtin(__builtin_amdgcn_s_wait_asynccnt)
  __builtin_amdgcn_s_wait_asynccnt(0);
#else
  asm volatile("s_wait_asynccnt 0" ::: "memory");
#endif
#endif
}

// Build a 16-f16 A-fragment from two contiguous 8-f16 chunks
// (callers pass p0 = row + half*16, p1 = p0 + 8: per lane, K = half*16 + e, contiguous).
__device__ __forceinline__ v16h frag_from_2x8(const _Float16* p0, const _Float16* p1) {
  v8h lo = *(const v8h*)p0;
  v8h hi = *(const v8h*)p1;
  v16h a;
#pragma unroll
  for (int i = 0; i < 8; ++i) { a[i] = lo[i]; a[i + 8] = hi[i]; }
  return a;
}

// ---------------- elementwise f32 -> f16 cast ----------------
__global__ void cast_f32_to_f16(const float* __restrict__ in,
                                _Float16* __restrict__ out, int n) {
  int i = blockIdx.x * blockDim.x + threadIdx.x;
  int stride = gridDim.x * blockDim.x;
  for (; i < n; i += stride) out[i] = (_Float16)in[i];
}

// ---------------- RoPE on head-major [b,h,s,d] f16 ----------------
__global__ void rope_f16(const _Float16* __restrict__ in,
                         _Float16* __restrict__ out,
                         const int* __restrict__ pos) {
  int idx = blockIdx.x * blockDim.x + threadIdx.x;
  const int NP = BATCH * NUM_HEADS * SEQ * (HEAD_DIM / 2);
  if (idx >= NP) return;
  int i  = idx & 31;          // rotation-pair index within head dim
  int rs = idx >> 5;          // (b*H+h)*SEQ + s
  int s  = rs & (SEQ - 1);
  float p    = (float)pos[s];
  float freq = __powf(10000.0f, -(float)i / 32.0f); // theta^(-2i/64)
  float sn, cs;
  __sincosf(p * freq, &sn, &cs);
  size_t base = (size_t)rs * HEAD_DIM + 2 * i;
  float x1 = (float)in[base];
  float x2 = (float)in[base + 1];
  out[base]     = (_Float16)(x1 * cs - x2 * sn);
  out[base + 1] = (_Float16)(x1 * sn + x2 * cs);
}

// ---------------- shared GEMM mainloop ----------------
// Computes acc[4] = X[m0:m0+16, :] * W[n0:n0+64, :]^T for this wave.
// B panel (64 x 32 f16 per K-step) is staged block-wide into LDS via the
// async-to-LDS path, double buffered; every wave of the block reads the same
// panel, cutting B global traffic 8x.
__device__ __forceinline__ void gemm_mainloop(const _Float16* __restrict__ X,
                                              const _Float16* __restrict__ W,
                                              _Float16* bs,   // LDS 2*64*32 f16
                                              int m0, int n0, int tid,
                                              v8f acc[4]) {
  const int lane = tid & 31;
  const int half = (lane >> 4) & 1;
  const int l16  = lane & 15;
  const int rr = tid >> 2;          // 0..63 : relative B row (output feature)
  const int c8 = (tid & 3) * 8;     // 0,8,16,24 : f16 column within panel

  const _Float16* arow = X + (size_t)(m0 + l16) * D_MODEL + half * 16;
  const _Float16* wrow = W + (size_t)(n0 + rr) * D_MODEL + c8;

  // prologue: stage K-step 0 into buffer 0
  stage16B(wrow, bs + rr * 32 + c8);

  for (int kk = 0; kk < D_MODEL; kk += 32) {
    const int cur = (kk >> 5) & 1;
    _Float16* bcur = bs + cur * (64 * 32);

    if (kk + 32 < D_MODEL) {
      stage16B(wrow + kk + 32, bs + (cur ^ 1) * (64 * 32) + rr * 32 + c8);
      wait_async_le1();              // current buffer's async retired (in-order)
    } else {
      wait_async_le0();
    }
    __syncthreads();                 // all waves' panel parts visible

    __builtin_prefetch(arow + kk + 256, 0, 1);
    v16h a = frag_from_2x8(arow + kk, arow + kk + 8);
#pragma unroll
    for (int nb = 0; nb < 4; ++nb) {
      // B-frag from LDS: lane holds column n, K = half*16 + e (contiguous 16 f16)
      v16h b = *(const v16h*)(bcur + (nb * 16 + l16) * 32 + half * 16);
      acc[nb] = __builtin_amdgcn_wmma_f32_16x16x32_f16(false, a, false, b,
                                                       (short)0, acc[nb], false, false);
    }
    __syncthreads();                 // protect bcur before it is re-staged
  }
}

// ---------------- projection GEMM:  out = X (T,1024) * W^T ----------------
// mode 0: write head-major [b,h,s,d]; mode 1: [b,h,d,s] (transposed V).
__global__ __launch_bounds__(256)
void proj_gemm_f16(const _Float16* __restrict__ X,
                   const _Float16* __restrict__ W,
                   _Float16* __restrict__ out, int mode) {
  __shared__ __align__(32) _Float16 bstage[2 * 64 * 32];
  const int tid  = threadIdx.x;
  const int lane = tid & 31;
  const int wave = tid >> 5;
  const int half = lane >> 4;
  const int l16  = lane & 15;
  const int m0 = blockIdx.y * 128 + wave * 16;
  const int n0 = blockIdx.x * 64;

  v8f acc[4] = {};
  gemm_mainloop(X, W, bstage, m0, n0, tid, acc);

#pragma unroll
  for (int nb = 0; nb < 4; ++nb) {
#pragma unroll
    for (int r = 0; r < 8; ++r) {
      int M = m0 + half * 8 + r;        // token t
      int N = n0 + nb * 16 + l16;       // feature o
      int b = M >> 11;                  // / SEQ
      int s = M & (SEQ - 1);
      int h = N >> 6;
      int d = N & 63;
      size_t idx = (mode == 0)
          ? ((size_t)((b * NUM_HEADS + h) * SEQ + s)) * HEAD_DIM + d
          : ((size_t)((b * NUM_HEADS + h) * HEAD_DIM + d)) * SEQ + s;
      out[idx] = (_Float16)acc[nb][r];
    }
  }
}

// ---------------- output GEMM: d_out = att (T,1024) * Wo^T, f32 store ----------------
__global__ __launch_bounds__(256)
void out_gemm_f32(const _Float16* __restrict__ X,
                  const _Float16* __restrict__ W,
                  float* __restrict__ out) {
  __shared__ __align__(32) _Float16 bstage[2 * 64 * 32];
  const int tid  = threadIdx.x;
  const int lane = tid & 31;
  const int wave = tid >> 5;
  const int half = lane >> 4;
  const int l16  = lane & 15;
  const int m0 = blockIdx.y * 128 + wave * 16;
  const int n0 = blockIdx.x * 64;

  v8f acc[4] = {};
  gemm_mainloop(X, W, bstage, m0, n0, tid, acc);

#pragma unroll
  for (int nb = 0; nb < 4; ++nb)
#pragma unroll
    for (int r = 0; r < 8; ++r) {
      int M = m0 + half * 8 + r;
      int N = n0 + nb * 16 + l16;
      out[(size_t)M * D_MODEL + N] = acc[nb][r];
    }
}

// ---------------- causal flash attention ----------------
// Q,K: [b,h,s,d] f16 (rope'd). Vt: [b,h,d,s] f16. att: [t, D_MODEL] f16.
__global__ __launch_bounds__(128)
void flash_attn_f16(const _Float16* __restrict__ Q,
                    const _Float16* __restrict__ Kc,
                    const _Float16* __restrict__ Vt,
                    _Float16* __restrict__ att) {
  __shared__ __align__(32) _Float16 pbuf[4][16 * 32];  // per-wave 16x32 P staging
  const int lane = threadIdx.x & 31;
  const int wave = threadIdx.x >> 5;
  const int half = lane >> 4;
  const int l16  = lane & 15;
  const int bh = blockIdx.y;
  const int bb = bh >> 4;
  const int hh = bh & 15;
  const int qt = blockIdx.x * 4 + wave;  // query tile (16 rows)
  const int q0 = qt * 16;

  const _Float16* Qb = Q  + (size_t)bh * SEQ * HEAD_DIM;
  const _Float16* Kb = Kc + (size_t)bh * SEQ * HEAD_DIM;
  const _Float16* Vb = Vt + (size_t)bh * HEAD_DIM * SEQ;

  const _Float16* qrow = Qb + (size_t)(q0 + l16) * HEAD_DIM + half * 16;
  v16h qa0 = frag_from_2x8(qrow,      qrow + 8);
  v16h qa1 = frag_from_2x8(qrow + 32, qrow + 40);

  float mrow[8], lrow[8];
#pragma unroll
  for (int r = 0; r < 8; ++r) { mrow[r] = -INFINITY; lrow[r] = 0.0f; }
  v8f o_acc[4] = {};
  _Float16* pl = &pbuf[wave][0];
  const float scale = 0.125f;  // 1/sqrt(64)

  for (int n0 = 0; n0 < q0 + 16; n0 += 32) {
    // --- scores: two 16x16 tiles covering kv columns [n0, n0+32)
    v8f s0 = {}, s1 = {};
    {
      const _Float16* k0 = Kb + (size_t)(n0 + l16) * HEAD_DIM + half * 16;
      v16h kb = *(const v16h*)k0;
      s0 = __builtin_amdgcn_wmma_f32_16x16x32_f16(false, qa0, false, kb, (short)0, s0, false, false);
      kb = *(const v16h*)(k0 + 32);
      s0 = __builtin_amdgcn_wmma_f32_16x16x32_f16(false, qa1, false, kb, (short)0, s0, false, false);
    }
    {
      const _Float16* k1 = Kb + (size_t)(n0 + 16 + l16) * HEAD_DIM + half * 16;
      v16h kb = *(const v16h*)k1;
      s1 = __builtin_amdgcn_wmma_f32_16x16x32_f16(false, qa0, false, kb, (short)0, s1, false, false);
      kb = *(const v16h*)(k1 + 32);
      s1 = __builtin_amdgcn_wmma_f32_16x16x32_f16(false, qa1, false, kb, (short)0, s1, false, false);
    }

    // --- causal mask, online softmax stats (16-lane column groups share a row)
    float p0[8], p1[8], alpha[8];
#pragma unroll
    for (int r = 0; r < 8; ++r) {
      int sq = q0 + half * 8 + r;
      float a0 = (n0 + l16      <= sq) ? s0[r] * scale : -INFINITY;
      float a1 = (n0 + 16 + l16 <= sq) ? s1[r] * scale : -INFINITY;
      float tm = fmaxf(a0, a1);
#pragma unroll
      for (int mk = 1; mk < 16; mk <<= 1) tm = fmaxf(tm, __shfl_xor(tm, mk, 32));
      float mnew = fmaxf(mrow[r], tm);          // finite: every row has >=1 valid col
      alpha[r] = __expf(mrow[r] - mnew);
      p0[r] = __expf(a0 - mnew);
      p1[r] = __expf(a1 - mnew);
      float rs = p0[r] + p1[r];
#pragma unroll
      for (int mk = 1; mk < 16; mk <<= 1) rs += __shfl_xor(rs, mk, 32);
      lrow[r] = lrow[r] * alpha[r] + rs;
      mrow[r] = mnew;
    }

#pragma unroll
    for (int g = 0; g < 4; ++g)
#pragma unroll
      for (int r = 0; r < 8; ++r) o_acc[g][r] *= alpha[r];

    // --- C-layout P -> LDS -> A-layout fragment (16x32 f16)
#pragma unroll
    for (int r = 0; r < 8; ++r) {
      pl[(half * 8 + r) * 32 + l16]      = (_Float16)p0[r];
      pl[(half * 8 + r) * 32 + 16 + l16] = (_Float16)p1[r];
    }
    asm volatile("s_wait_dscnt 0" ::: "memory");
    v16h pa = frag_from_2x8(pl + l16 * 32 + half * 16,
                            pl + l16 * 32 + half * 16 + 8);

    // --- O += P @ V   (V transposed: per-lane column d, contiguous in s)
#pragma unroll
    for (int g = 0; g < 4; ++g) {
      const _Float16* vp = Vb + (size_t)(g * 16 + l16) * SEQ + n0 + half * 16;
      v16h vb = *(const v16h*)vp;
      o_acc[g] = __builtin_amdgcn_wmma_f32_16x16x32_f16(false, pa, false, vb,
                                                        (short)0, o_acc[g], false, false);
    }
  }

  // --- normalize and store as [t, D_MODEL] for the output projection
#pragma unroll
  for (int g = 0; g < 4; ++g)
#pragma unroll
    for (int r = 0; r < 8; ++r) {
      int sq = q0 + half * 8 + r;
      size_t t = (size_t)bb * SEQ + sq;
      int o = hh * HEAD_DIM + g * 16 + l16;
      att[t * D_MODEL + o] = (_Float16)(o_acc[g][r] / lrow[r]);
    }
}

extern "C" void kernel_launch(void* const* d_in, const int* in_sizes, int n_in,
                              void* d_out, int out_size, void* d_ws, size_t ws_size,
                              hipStream_t stream) {
  const float* x   = (const float*)d_in[0];
  const int*   pos = (const int*)d_in[1];
  const float* wq  = (const float*)d_in[2];
  const float* wk  = (const float*)d_in[3];
  const float* wv  = (const float*)d_in[4];
  const float* wo  = (const float*)d_in[5];
  float* out = (float*)d_out;

  char* ws = (char*)d_ws;
  const size_t MB = 1024 * 1024;
  _Float16* xh  = (_Float16*)(ws +   0 * MB);  // 16 MiB  x in f16
  _Float16* wqh = (_Float16*)(ws +  16 * MB);  //  2 MiB each
  _Float16* wkh = (_Float16*)(ws +  18 * MB);
  _Float16* wvh = (_Float16*)(ws +  20 * MB);
  _Float16* woh = (_Float16*)(ws +  22 * MB);
  _Float16* qhd = (_Float16*)(ws +  24 * MB);  // 16 MiB  Q [b,h,s,d]
  _Float16* khd = (_Float16*)(ws +  40 * MB);  // 16 MiB  K [b,h,s,d]
  _Float16* vt  = (_Float16*)(ws +  56 * MB);  // 16 MiB  V [b,h,d,s]
  _Float16* qr  = (_Float16*)(ws +  72 * MB);  // 16 MiB  rope(Q)
  _Float16* kr  = (_Float16*)(ws +  88 * MB);  // 16 MiB  rope(K)
  _Float16* att = (_Float16*)(ws + 104 * MB);  // 16 MiB  attention out [t,D]

  const int nX = T_TOTAL * D_MODEL;
  const int nW = D_MODEL * D_MODEL;
  cast_f32_to_f16<<<1024, 256, 0, stream>>>(x,  xh,  nX);
  cast_f32_to_f16<<<512,  256, 0, stream>>>(wq, wqh, nW);
  cast_f32_to_f16<<<512,  256, 0, stream>>>(wk, wkh, nW);
  cast_f32_to_f16<<<512,  256, 0, stream>>>(wv, wvh, nW);
  cast_f32_to_f16<<<512,  256, 0, stream>>>(wo, woh, nW);

  dim3 gp(D_MODEL / 64, T_TOTAL / 128);
  proj_gemm_f16<<<gp, 256, 0, stream>>>(xh, wqh, qhd, 0);
  proj_gemm_f16<<<gp, 256, 0, stream>>>(xh, wkh, khd, 0);
  proj_gemm_f16<<<gp, 256, 0, stream>>>(xh, wvh, vt,  1);

  const int npairs = BATCH * NUM_HEADS * SEQ * (HEAD_DIM / 2);
  rope_f16<<<(npairs + 255) / 256, 256, 0, stream>>>(qhd, qr, pos);
  rope_f16<<<(npairs + 255) / 256, 256, 0, stream>>>(khd, kr, pos);

  dim3 ga(SEQ / 16 / 4, BATCH * NUM_HEADS);
  flash_attn_f16<<<ga, 128, 0, stream>>>(qr, kr, vt, att);

  out_gemm_f32<<<gp, 256, 0, stream>>>(att, woh, out);

  (void)in_sizes; (void)n_in; (void)out_size; (void)ws_size;
}
